// FastWAVLayer_47579647705724
// MI455X (gfx1250) — matrix operand order repaired and checked by reading, not compiled, not run on hardware
//
#include <hip/hip_runtime.h>

typedef _Float16 v16h __attribute__((ext_vector_type(16)));
typedef _Float16 v8h  __attribute__((ext_vector_type(8)));
typedef float    v8f  __attribute__((ext_vector_type(8)));

#define IN_DIM  1024
#define OUT_DIM 1024
#define BATCH   256

// -------- prep 1: sxh[b,i] = (f16) silu(x[b,i]), row-major --------
__global__ void silu_f16_kernel(const float* __restrict__ x,
                                _Float16* __restrict__ sxh, int n) {
  int t = blockIdx.x * blockDim.x + threadIdx.x;
  if (t < n) {
    float v = x[t];
    // silu(v) = v / (1 + exp(-v)); exp(-v) = exp2(-v * log2(e))
    float e = __builtin_amdgcn_exp2f(v * -1.4426950408889634f);
    sxh[t] = (_Float16)(v / (1.0f + e));
  }
}

// -------- prep 2: bwt[o,i] = (f16) base_weight[i,o] (LDS-tiled transpose) ----
__global__ void transpose_f16_kernel(const float* __restrict__ w,
                                     _Float16* __restrict__ wt) {
  __shared__ _Float16 tile[32][33];
  const int o0 = blockIdx.x * 32;
  const int i0 = blockIdx.y * 32;
  const int tx = threadIdx.x & 31;   // inner dim
  const int ty = threadIdx.x >> 5;   // 0..7
#pragma unroll
  for (int rr = 0; rr < 4; ++rr) {
    int il = ty + rr * 8;
    tile[il][tx] = (_Float16)w[(size_t)(i0 + il) * OUT_DIM + (o0 + tx)];
  }
  __syncthreads();
#pragma unroll
  for (int rr = 0; rr < 4; ++rr) {
    int ol = ty + rr * 8;
    wt[(size_t)(o0 + ol) * IN_DIM + (i0 + tx)] = tile[tx][ol];
  }
}

// -------- main fused kernel: base GEMM via WMMA + wavelet via VALU ----------
// block = 128 threads (4 waves). Block tile: 16 b x 64 o. Wave w owns o-strip
// [o_tile + 16w, +16). C-fragment mapping (ISA 7.12.2): lane l, VGPR r holds
// element (m = r + 8*(l/16), n = l%16). Wavelet accumulators use the same map.
// The block's full x tile (16 x 1024 f32 = 64KB) is staged into LDS ONCE, so
// the K-loop runs with no barriers; xs reads are lane-uniform broadcasts.
__global__ void __launch_bounds__(128)
fastwav_kernel(const float* __restrict__ x,
               const float* __restrict__ scale,
               const float* __restrict__ translation,
               const float* __restrict__ wweight,
               const _Float16* __restrict__ sxh,
               const _Float16* __restrict__ bwt,
               float* __restrict__ out) {
  __shared__ __align__(16) float xs[16 * IN_DIM];   // 64 KB of 320 KB/WGP

  const int tid  = threadIdx.x;
  const int wave = tid >> 5;
  const int lane = tid & 31;
  const int hw   = lane >> 4;   // half-wave: 0/1
  const int l16  = lane & 15;

  const int b_base = blockIdx.y * 16;
  const int o_lane = blockIdx.x * 64 + wave * 16 + l16;

  // ---- one-shot cooperative stage of x[16][1024] into LDS (float4) ----
#pragma unroll 4
  for (int c = 0; c < 32; ++c) {
    int f = tid * 4 + c * 512;           // 0..16380, 16B aligned
    int row = f >> 10, col = f & 1023;
    *(float4*)(xs + f) =
        *(const float4*)(x + (size_t)(b_base + row) * IN_DIM + col);
  }
  __syncthreads();

  v8f cacc = {};
  float wacc[8];
#pragma unroll
  for (int r = 0; r < 8; ++r) wacc[r] = 0.0f;

  const _Float16* aRow = sxh + (size_t)(b_base + l16) * IN_DIM;  // A row m=l16
  const _Float16* bRow = bwt + (size_t)o_lane * IN_DIM;          // B col n=l16
  const float* xRow0 = xs + (size_t)(hw * 8) * IN_DIM;           // lane's b-rows

  for (int kk = 0; kk < IN_DIM; kk += 32) {
    // ---- A fragment: lane holds K = hw*8+0..7 and 16+hw*8+0..7 of row m ----
    union { v16h v; v8h h[2]; } A;
    A.h[0] = *(const v8h*)(aRow + kk + hw * 8);
    A.h[1] = *(const v8h*)(aRow + kk + 16 + hw * 8);
    // ---- B fragment: lane holds K = hw*8+0..7 and 16+hw*8+0..7 of column n ----
    union { v16h v; v8h h[2]; } B;
    B.h[0] = *(const v8h*)(bRow + kk + hw * 8);
    B.h[1] = *(const v8h*)(bRow + kk + 16 + hw * 8);

    cacc = __builtin_amdgcn_wmma_f32_16x16x32_f16(
        false, A.v, false, B.v, (short)0, cacc, false, false);

    // ---- wavelet path: this lane handles o = o_lane, b = b_base+hw*8+r ----
    const float* sP = scale       + (size_t)kk * OUT_DIM + o_lane;
    const float* tP = translation + (size_t)kk * OUT_DIM + o_lane;
    const float* wP = wweight     + (size_t)kk * OUT_DIM + o_lane;
    __builtin_prefetch(sP + (size_t)32 * OUT_DIM, 0, 1);
    __builtin_prefetch(tP + (size_t)32 * OUT_DIM, 0, 1);
    __builtin_prefetch(wP + (size_t)32 * OUT_DIM, 0, 1);
#pragma unroll 4
    for (int j = 0; j < 32; ++j) {
      float s  = sP[(size_t)j * OUT_DIM];
      float tr = tP[(size_t)j * OUT_DIM];
      float w  = wP[(size_t)j * OUT_DIM];
      // v_rcp_f32 (~1 ulp) instead of a ~10-op IEEE divide; scale ~ 1.0
      float rs = __builtin_amdgcn_rcpf(s + 1e-8f);
#pragma unroll
      for (int r = 0; r < 8; ++r) {
        float xv = xRow0[(size_t)r * IN_DIM + kk + j];   // LDS broadcast read
        float t  = (xv - tr) * rs;
        float t2 = t * t;
        // exp(-0.5*t2) = exp2(t2 * (-0.5*log2 e))
        float e  = __builtin_amdgcn_exp2f(t2 * -0.7213475204444817f);
        wacc[r] = fmaf((1.0f - t2) * e, w, wacc[r]);
      }
    }
  }

  // ---- store: C-frag element r lives at (b_base + r + 8*hw, o_lane) ----
#pragma unroll
  for (int r = 0; r < 8; ++r) {
    int b = b_base + r + hw * 8;
    out[(size_t)b * OUT_DIM + o_lane] = cacc[r] + wacc[r];
  }
}

extern "C" void kernel_launch(void* const* d_in, const int* in_sizes, int n_in,
                              void* d_out, int out_size, void* d_ws, size_t ws_size,
                              hipStream_t stream) {
  (void)in_sizes; (void)n_in; (void)out_size; (void)ws_size;
  const float* x           = (const float*)d_in[0];
  const float* scale       = (const float*)d_in[1];
  const float* translation = (const float*)d_in[2];
  const float* wweight     = (const float*)d_in[3];
  const float* bweight     = (const float*)d_in[4];
  float* out = (float*)d_out;

  _Float16* sxh = (_Float16*)d_ws;                         // 512 KB
  _Float16* bwt = (_Float16*)((char*)d_ws +
                  (size_t)BATCH * IN_DIM * sizeof(_Float16)); // 2 MB

  const int n = BATCH * IN_DIM;
  hipLaunchKernelGGL(silu_f16_kernel, dim3((n + 255) / 256), dim3(256), 0,
                     stream, x, sxh, n);
  hipLaunchKernelGGL(transpose_f16_kernel, dim3(OUT_DIM / 32, IN_DIM / 32),
                     dim3(256), 0, stream, bweight, bwt);
  hipLaunchKernelGGL(fastwav_kernel, dim3(OUT_DIM / 64, BATCH / 16),
                     dim3(128), 0, stream,
                     x, scale, translation, wweight, sxh, bwt, out);
}
